// LinearMemoryAttention_68539088109655
// MI455X (gfx1250) — matrix operand order repaired and checked by reading, not compiled, not run on hardware
//
#include <hip/hip_runtime.h>

// ---------------------------------------------------------------------------
// Chunked linear attention with persistent tensor-product memory (MI455X).
// B=1, S=4096, D=P=512, H=8, Dh=64, chunk C=64, NC=64.
// GEMMs: v_wmma_f32_16x16x32_f16. Staging: global_load_async_to_lds_b128.
// All WMMA fragments are fetched as contiguous b128-class vector loads by
// storing each staged tile in the orientation its fragment pattern wants.
// ---------------------------------------------------------------------------

typedef __attribute__((ext_vector_type(16))) _Float16 v16h;
typedef __attribute__((ext_vector_type(8)))  _Float16 v8h;
typedef __attribute__((ext_vector_type(8)))  float    v8f;

#define S_LEN 4096
#define DMODEL 512
#define NHEADS 8
#define DHEAD 64
#define CHUNK 64
#define NCHUNK (S_LEN / CHUNK)

static __device__ __forceinline__ v8f wmma16(v16h a, v16h b, v8f c) {
  return __builtin_amdgcn_wmma_f32_16x16x32_f16(
      false, a, false, b, (short)0, c, false, false);
}

// Fragment K order used here for v_wmma_f32_16x16x32_f16 (contiguous, same for A and B):
//  A (16x32): lane<16 K=0..15, lane>=16 K=16..31    -> two 8-half runs @ 2*kb8, 2*kb8+8
//  B (32x16): lane<16 K=0..15, lane>=16 K=16..31    -> one 16-half run @ kb16
//  C/D (16x16 f32): n = lane&15, m = j + 8*(lane>>4)

static __device__ __forceinline__ v16h mk_a(const _Float16* row, int kb8) {
  v8h lo = *(const v8h*)(row + 2 * kb8);
  v8h hv = *(const v8h*)(row + 2 * kb8 + 8);
  v16h a;
#pragma unroll
  for (int i = 0; i < 8; ++i) { a[i] = lo[i]; a[i + 8] = hv[i]; }
  return a;
}

static __device__ __forceinline__ float elu1(float x) {
  float e = __expf(x);
  return (x > 0.0f) ? (x + 1.0f) : e;
}

// --- CDNA5 async global->LDS staging (ASYNCcnt-tracked) --------------------
static __device__ __forceinline__ void async_ld_b128(void* lds_dst, const void* gsrc) {
  unsigned loff = (unsigned)(uintptr_t)lds_dst;  // LDS aperture: offset in addr[31:0]
  asm volatile("global_load_async_to_lds_b128 %0, %1, off"
               :: "v"(loff), "v"(gsrc) : "memory");
}
static __device__ __forceinline__ void async_wait0() {
  asm volatile("s_wait_asynccnt 0" ::: "memory");
}

// ---------------------------------------------------------------------------
// 4096x512x512 GEMM, block tile 128x64, wave tile 16x64 (4 WMMA/K-step).
// MODE 0: f32 in, +bias, ELU+1, f16 out | MODE 1: +bias, f16 out
// MODE 2: f16 in, no bias, f32 out
// ---------------------------------------------------------------------------
template <int MODE>
__global__ __launch_bounds__(256) void k_gemm(const void* __restrict__ Xv,
                                              const float* __restrict__ W,
                                              const float* __restrict__ bias,
                                              void* __restrict__ Yv) {
  __shared__ __attribute__((aligned(32))) _Float16 BtT[64 * 32];  // [n][k]
  const int wave = threadIdx.x >> 5, lane = threadIdx.x & 31;
  const int l15 = lane & 15, hi = (lane >> 4) & 1;
  const int m0 = blockIdx.x * 128 + wave * 16;
  const int n0 = blockIdx.y * 64;
  const int mrow = m0 + l15;
  const int kb8 = hi ? 8 : 0, kb16 = hi ? 16 : 0;
  const float*    Xf = (const float*)Xv;
  const _Float16* Xh = (const _Float16*)Xv;
  v8f acc[4] = {};

  for (int k0 = 0; k0 < DMODEL; k0 += 32) {
    {  // stage 32x64 weight tile transposed so B fragments are contiguous
      const int k = threadIdx.x >> 3;        // 0..31
      const int n8 = (threadIdx.x & 7) * 8;  // 0..56
      const float* wp = W + (size_t)(k0 + k) * DMODEL + n0 + n8;
#pragma unroll
      for (int j = 0; j < 8; ++j) BtT[(n8 + j) * 32 + k] = (_Float16)wp[j];
    }
    __syncthreads();

    v16h a;
    if constexpr (MODE == 2) {
      const _Float16* xp = Xh + (size_t)mrow * DMODEL + k0;
      a = mk_a(xp, kb8);
      if (k0 + 64 < DMODEL) __builtin_prefetch(xp + 64, 0, 1);
    } else {
      const float* xp = Xf + (size_t)mrow * DMODEL + k0;
#pragma unroll
      for (int i = 0; i < 8; ++i) {
        a[i]     = (_Float16)xp[2 * kb8 + i];
        a[i + 8] = (_Float16)xp[2 * kb8 + 8 + i];
      }
      if (k0 + 64 < DMODEL) __builtin_prefetch(xp + 64, 0, 1);
    }
#pragma unroll
    for (int c = 0; c < 4; ++c) {
      v16h b = *(const v16h*)(&BtT[(c * 16 + l15) * 32 + kb16]);
      acc[c] = wmma16(a, b, acc[c]);
    }
    __syncthreads();
  }

#pragma unroll
  for (int c = 0; c < 4; ++c) {
    const int n = n0 + c * 16 + l15;
    const float bv = (MODE == 2) ? 0.0f : bias[n];
#pragma unroll
    for (int j = 0; j < 8; ++j) {
      const int m = m0 + j + hi * 8;
      float y = acc[c][j] + bv;
      if constexpr (MODE == 0) y = elu1(y);
      if constexpr (MODE == 2) ((float*)Yv)[(size_t)m * DMODEL + n] = y;
      else ((_Float16*)Yv)[(size_t)m * DMODEL + n] = (_Float16)y;
    }
  }
}

// ---------------------------------------------------------------------------
// Per (head h, chunk c): G^T[e,d] = sum_t sk[t,d]*v[t,e] (WMMA over time),
// g[d] = sum_t sk[t,d].  sk/v staged TRANSPOSED ([d][t], [e][t]) so every
// fragment is a contiguous LDS vector load.
// ---------------------------------------------------------------------------
__global__ __launch_bounds__(256) void k_chunk_sums(const _Float16* __restrict__ skb,
                                                    const _Float16* __restrict__ vb,
                                                    float* __restrict__ GT,
                                                    float* __restrict__ g) {
  __shared__ __attribute__((aligned(32))) _Float16 SKtT[DHEAD * CHUNK];  // [d][t]
  __shared__ __attribute__((aligned(32))) _Float16 VtT[DHEAD * CHUNK];   // [e][t]
  const int h = blockIdx.x >> 6, c = blockIdx.x & (NCHUNK - 1);
  const int t0 = c * CHUNK, col = h * DHEAD;
  const int wave = threadIdx.x >> 5, lane = threadIdx.x & 31;
  const int l15 = lane & 15, hi = (lane >> 4) & 1;
  const int kb8 = hi ? 8 : 0, kb16 = hi ? 16 : 0;

#pragma unroll
  for (int r = 0; r < 2; ++r) {                    // transpose-stage both tiles
    const int idx = threadIdx.x + r * 256;         // 512 b128 segments / tile
    const int trow = idx >> 3, dseg = (idx & 7) * 8;
    const size_t go = (size_t)(t0 + trow) * DMODEL + col + dseg;
    v8h ks = *(const v8h*)(skb + go);
    v8h kv = *(const v8h*)(vb + go);
#pragma unroll
    for (int j = 0; j < 8; ++j) {
      SKtT[(dseg + j) * CHUNK + trow] = ks[j];
      VtT[(dseg + j) * CHUNK + trow] = kv[j];
    }
  }
  __syncthreads();

  float* Gt = GT + (size_t)blockIdx.x * (DHEAD * DHEAD);
  for (int s = wave; s < 16; s += 8) {
    const int d0 = (s >> 2) << 4, e0 = (s & 3) << 4;
    v8f acc = {};
    for (int ks = 0; ks < CHUNK; ks += 32) {
      v16h a = mk_a(&SKtT[(d0 + l15) * CHUNK + ks], kb8);
      v16h b = *(const v16h*)(&VtT[(e0 + l15) * CHUNK + ks + kb16]);
      acc = wmma16(a, b, acc);
    }
    const int n = e0 + l15;  // e index
#pragma unroll
    for (int j = 0; j < 8; ++j)              // store TRANSPOSED: GT[e][d]
      Gt[n * DHEAD + (d0 + j + hi * 8)] = acc[j];
  }
  if (threadIdx.x < DHEAD) {                 // g[d] = contiguous row sum
    float acc = 0.0f;
    for (int t = 0; t < CHUNK; ++t) acc += (float)SKtT[threadIdx.x * CHUNK + t];
    g[(size_t)blockIdx.x * DHEAD + threadIdx.x] = acc;
  }
}

// ---------------------------------------------------------------------------
// Exclusive prefix over chunk states ([e][d] layout), f32 accumulate,
// f16 state output (it is a WMMA operand downstream).
// ---------------------------------------------------------------------------
__global__ __launch_bounds__(256) void k_prefix(const float* __restrict__ GT,
                                                const float* __restrict__ g,
                                                const float* __restrict__ M_mem,
                                                const float* __restrict__ z_mem,
                                                _Float16* __restrict__ stateT,
                                                float* __restrict__ zstate) {
  const int h = blockIdx.x;
  const int base = threadIdx.x * 16;
  const int e = base >> 6, d = base & 63;  // [e][d]: M_mem is [d][e] -> transpose
  float acc[16];
#pragma unroll
  for (int i = 0; i < 16; ++i)
    acc[i] = M_mem[(size_t)h * DHEAD * DHEAD + (size_t)(d + i) * DHEAD + e];
  float zacc = (threadIdx.x < DHEAD) ? z_mem[h * DHEAD + threadIdx.x] : 0.0f;

  for (int c = 0; c < NCHUNK; ++c) {
    const size_t off = ((size_t)h * NCHUNK + c) * (DHEAD * DHEAD);
#pragma unroll
    for (int i = 0; i < 16; ++i) {
      stateT[off + base + i] = (_Float16)acc[i];
      acc[i] += GT[off + base + i];
    }
    if (threadIdx.x < DHEAD) {
      const size_t zo = ((size_t)h * NCHUNK + c) * DHEAD + threadIdx.x;
      zstate[zo] = zacc;
      zacc += g[zo];
    }
  }
}

// ---------------------------------------------------------------------------
// Per (head, chunk): scores = SQ@SK^T -> causal mask -> N = SQ@state^T +
// tril(S)@V ; den = rowsum(tril) + sq.z + EPS ; O = N/den (f16).
// sq/sk async-staged row-major (scores want rows); v transpose-staged [e][t];
// state^T f16 B fragments come straight from global as v16h.
// ---------------------------------------------------------------------------
__global__ __launch_bounds__(256) void k_chunk_attn(const _Float16* __restrict__ sqb,
                                                    const _Float16* __restrict__ skb,
                                                    const _Float16* __restrict__ vb,
                                                    const _Float16* __restrict__ stateT,
                                                    const float* __restrict__ zstate,
                                                    _Float16* __restrict__ ob) {
  __shared__ __attribute__((aligned(32))) _Float16 SQt[CHUNK * DHEAD];  // [t][d]
  __shared__ __attribute__((aligned(32))) _Float16 SKt[CHUNK * DHEAD];  // [t][d]
  __shared__ __attribute__((aligned(32))) _Float16 VtT[DHEAD * CHUNK];  // [e][t]
  __shared__ __attribute__((aligned(32))) _Float16 S16[CHUNK * CHUNK];  // [t][t']
  __shared__ float S32[CHUNK * CHUNK];
  __shared__ float den[CHUNK];
  const int h = blockIdx.x >> 6, c = blockIdx.x & (NCHUNK - 1);
  const int t0 = c * CHUNK, col = h * DHEAD;
  const int wave = threadIdx.x >> 5, lane = threadIdx.x & 31;
  const int l15 = lane & 15, hi = (lane >> 4) & 1;
  const int kb8 = hi ? 8 : 0, kb16 = hi ? 16 : 0;

  // issue async row-major staging first ...
#pragma unroll
  for (int r = 0; r < 2; ++r) {
    const int idx = threadIdx.x + r * 256;
    const int row = idx >> 3, seg = (idx & 7) * 8;
    const size_t go = (size_t)(t0 + row) * DMODEL + col + seg;
    const int lo = row * DHEAD + seg;
    async_ld_b128(&SQt[lo], sqb + go);
    async_ld_b128(&SKt[lo], skb + go);
  }
  // ... and overlap the V transpose staging under the ASYNCcnt shadow
#pragma unroll
  for (int r = 0; r < 2; ++r) {
    const int idx = threadIdx.x + r * 256;
    const int trow = idx >> 3, eseg = (idx & 7) * 8;
    v8h kv = *(const v8h*)(vb + (size_t)(t0 + trow) * DMODEL + col + eseg);
#pragma unroll
    for (int j = 0; j < 8; ++j) VtT[(eseg + j) * CHUNK + trow] = kv[j];
  }
  async_wait0();
  __syncthreads();

  // scores[t,t'] = sum_d sq[t,d]*sk[t',d]
  for (int s = wave; s < 16; s += 8) {
    const int m0 = (s >> 2) << 4, n0 = (s & 3) << 4;
    v8f acc = {};
    for (int ks = 0; ks < DHEAD; ks += 32) {
      v16h a = mk_a(&SQt[(m0 + l15) * DHEAD + ks], kb8);
      v16h b = *(const v16h*)(&SKt[(n0 + l15) * DHEAD + ks + kb16]);
      acc = wmma16(a, b, acc);
    }
#pragma unroll
    for (int j = 0; j < 8; ++j) {
      const int m = m0 + j + hi * 8, n = n0 + l15;
      float val = (n <= m) ? acc[j] : 0.0f;  // causal, diagonal inclusive
      S32[m * CHUNK + n] = val;
      S16[m * CHUNK + n] = (_Float16)val;
    }
  }
  __syncthreads();

  if (threadIdx.x < CHUNK) {
    const int t = threadIdx.x;
    float rs = 0.0f;
    for (int j = 0; j < CHUNK; ++j) rs += S32[t * CHUNK + j];
    const float* zs = zstate + (size_t)blockIdx.x * DHEAD;
    float dot = 0.0f;
    for (int d = 0; d < DHEAD; ++d) dot += (float)SQt[t * DHEAD + d] * zs[d];
    den[t] = rs + dot + 1e-6f;
  }
  __syncthreads();

  // N = SQ @ state^T  +  tril(scores) @ V
  const _Float16* st = stateT + (size_t)blockIdx.x * (DHEAD * DHEAD);  // [e][d]
  for (int s = wave; s < 16; s += 8) {
    const int m0 = (s >> 2) << 4, e0 = (s & 3) << 4;
    v8f acc = {};
    for (int ks = 0; ks < DHEAD; ks += 32) {
      v16h a  = mk_a(&SQt[(m0 + l15) * DHEAD + ks], kb8);
      v16h b  = *(const v16h*)(st + (e0 + l15) * DHEAD + ks + kb16);
      v16h a2 = mk_a(&S16[(m0 + l15) * CHUNK + ks], kb8);
      v16h b2 = *(const v16h*)(&VtT[(e0 + l15) * CHUNK + ks + kb16]);
      acc = wmma16(a, b, acc);
      acc = wmma16(a2, b2, acc);
    }
#pragma unroll
    for (int j = 0; j < 8; ++j) {
      const int m = m0 + j + hi * 8, n = e0 + l15;
      ob[(size_t)(t0 + m) * DMODEL + col + n] = (_Float16)(acc[j] / den[m]);
    }
  }
}

// ---------------------------------------------------------------------------
extern "C" void kernel_launch(void* const* d_in, const int* in_sizes, int n_in,
                              void* d_out, int out_size, void* d_ws, size_t ws_size,
                              hipStream_t stream) {
  const float* X    = (const float*)d_in[0];
  const float* Wq   = (const float*)d_in[1];
  const float* bq   = (const float*)d_in[2];
  const float* Wk   = (const float*)d_in[3];
  const float* bk   = (const float*)d_in[4];
  const float* Wv   = (const float*)d_in[5];
  const float* bv   = (const float*)d_in[6];
  const float* Wo   = (const float*)d_in[7];
  const float* Mmem = (const float*)d_in[8];
  const float* zmem = (const float*)d_in[9];

  _Float16* sq = (_Float16*)d_ws;
  _Float16* sk = sq + (size_t)S_LEN * DMODEL;
  _Float16* v  = sk + (size_t)S_LEN * DMODEL;
  _Float16* ob = v  + (size_t)S_LEN * DMODEL;
  _Float16* stateT = ob + (size_t)S_LEN * DMODEL;               // f16 [h,c][e][d]
  float* GT     = (float*)(stateT + (size_t)NHEADS * NCHUNK * DHEAD * DHEAD);
  float* gsum   = GT   + (size_t)NHEADS * NCHUNK * DHEAD * DHEAD;
  float* zstate = gsum + (size_t)NHEADS * NCHUNK * DHEAD;

  dim3 gGemm(S_LEN / 128, DMODEL / 64), bGemm(256);
  k_gemm<0><<<gGemm, bGemm, 0, stream>>>(X, Wq, bq, sq);
  k_gemm<0><<<gGemm, bGemm, 0, stream>>>(X, Wk, bk, sk);
  k_gemm<1><<<gGemm, bGemm, 0, stream>>>(X, Wv, bv, v);

  k_chunk_sums<<<NHEADS * NCHUNK, 256, 0, stream>>>(sk, v, GT, gsum);
  k_prefix<<<NHEADS, 256, 0, stream>>>(GT, gsum, Mmem, zmem, stateT, zstate);
  k_chunk_attn<<<NHEADS * NCHUNK, 256, 0, stream>>>(sq, sk, v, stateT, zstate, ob);

  k_gemm<2><<<gGemm, bGemm, 0, stream>>>(ob, Wo, nullptr, (float*)d_out);
}
